// ReformerLayer_34634616275460
// MI455X (gfx1250) — matrix operand order repaired and checked by reading, not compiled, not run on hardware
//
#include <hip/hip_runtime.h>
#include <hip/hip_bf16.h>
#include <math.h>

// Problem constants (from reference)
#define B_      2
#define N_      4096
#define C_      1024
#define HEADS_  16
#define DH_     64
#define NHASH_  4
#define BUCKET_ 64
#define T_      N_
#define L_      (NHASH_ * T_)        // 16384 slots per bh
#define NBUCK_  (T_ / BUCKET_)       // 64 buckets per hash
#define NCHUNK_ (NHASH_ * NBUCK_)    // 256 chunks per bh
#define ROWS_   (B_ * N_)            // 8192
#define BH_     (B_ * HEADS_)        // 32

typedef __attribute__((ext_vector_type(16))) __bf16 bf16x16;
typedef __attribute__((ext_vector_type(8)))  float  f32x8;
typedef __attribute__((ext_vector_type(4)))  unsigned int u32x4;
typedef __attribute__((ext_vector_type(8)))  int i32x8;
typedef __attribute__((ext_vector_type(4)))  int i32x4;

union Bf16Frag { bf16x16 v; uint4 q[2]; };

#if __has_builtin(__builtin_amdgcn_tensor_load_to_lds) && \
    __has_builtin(__builtin_amdgcn_s_wait_tensorcnt)
#define USE_TDM 1
#else
#define USE_TDM 0
#endif

// A-matrix 16x32 bf16 fragment from row-major tile (ld elems/row).
// lane<16: row=lane, K{0..7,16..23}; lane>=16: row=lane-16, K{8..15,24..31}.
__device__ __forceinline__ bf16x16 load_frag_a(const __bf16* tile, int ld) {
  const int lane = threadIdx.x & 31;
  const int row = lane & 15, hf = lane >> 4;
  const __bf16* p = tile + row * ld + hf * 8;
  Bf16Frag u;
  u.q[0] = *(const uint4*)(p);
  u.q[1] = *(const uint4*)(p + 16);
  return u.v;
}

// B-matrix 32x16 bf16 fragment, source tile stored N-major (row=n, ld elems/row).
// lane<16: col=lane, K{0..7,16..23}; lane>=16: col=lane-16, K{8..15,24..31}.
__device__ __forceinline__ bf16x16 load_frag_bT(const __bf16* tileT, int ld) {
  const int lane = threadIdx.x & 31;
  const int col = lane & 15, hf = lane >> 4;
  const __bf16* p = tileT + col * ld + hf * 8;
  Bf16Frag u;
  u.q[0] = *(const uint4*)(p);
  u.q[1] = *(const uint4*)(p + 16);
  return u.v;
}

#if USE_TDM
// Workgroup-relative LDS byte address (low 32 bits of the generic pointer).
__device__ __forceinline__ unsigned lds_off(const void* p) {
  return (unsigned)(uintptr_t)p;
}

// Issue TENSOR_LOAD_TO_LDS of a 2-D bf16 tile: tile_d0 contiguous elements x
// tile_d1 rows, row stride = stride_elems, into LDS (packed, row-major).
// D# layout per CDNA5 ISA.
__device__ __forceinline__ void tdm_load_tile_bf16(
    const void* gaddr, unsigned lds_byte_addr,
    unsigned tile_d0, unsigned tile_d1, unsigned long long stride_elems) {
  const unsigned long long ga = (unsigned long long)gaddr;
  u32x4 g0;
  g0[0] = 1u;                                        // count=1, user descriptor
  g0[1] = lds_byte_addr;                             // lds_addr
  g0[2] = (unsigned)(ga & 0xffffffffull);            // global_addr[31:0]
  g0[3] = (unsigned)((ga >> 32) & 0x1ffffffull)      // global_addr[56:32]
        | 0x80000000u;                               // type=2 ("image")
  const unsigned long long td0 = stride_elems;       // tensor_dim0 (no OOB hit)
  const unsigned long long td1 = tile_d1;            // tensor_dim1
  // group1 (256 bits) as four 64-bit words
  unsigned long long w0 = (1ull << 16)               // data_size=1 -> 2 bytes
                        | ((td0 & 0xffffull) << 48); // tensor_dim0[15:0]
  unsigned long long w1 = ((td0 >> 16) & 0xffffull)  // tensor_dim0[31:16]
                        | ((td1 & 0xffffffffull) << 16)        // tensor_dim1
                        | (((unsigned long long)tile_d0 & 0xffffull) << 48);
  unsigned long long w2 = ((unsigned long long)tile_d1 & 0xffffull) // tile_dim1
                        | ((stride_elems & 0xffffffffull) << 32);   // stride[31:0]
  unsigned long long w3 = (stride_elems >> 32) & 0xffffull;         // stride[47:32]
  i32x8 g1;
  g1[0] = (int)(unsigned)(w0 & 0xffffffffull); g1[1] = (int)(unsigned)(w0 >> 32);
  g1[2] = (int)(unsigned)(w1 & 0xffffffffull); g1[3] = (int)(unsigned)(w1 >> 32);
  g1[4] = (int)(unsigned)(w2 & 0xffffffffull); g1[5] = (int)(unsigned)(w2 >> 32);
  g1[6] = (int)(unsigned)(w3 & 0xffffffffull); g1[7] = (int)(unsigned)(w3 >> 32);
  i32x4 z4 = {0, 0, 0, 0};                           // 2-D: tile_dim3/4 = 0
#if defined(__clang_major__) && (__clang_major__ >= 23)
  i32x8 z8 = {};
  __builtin_amdgcn_tensor_load_to_lds(g0, g1, z4, z4, z8, 0);
#else
  __builtin_amdgcn_tensor_load_to_lds(g0, g1, z4, z4, 0);
#endif
}
#endif  // USE_TDM

// ---------------------------------------------------------------- f32 -> bf16
__global__ void f2bf_kernel(const float* __restrict__ in, __bf16* __restrict__ out, size_t n) {
  size_t i = (size_t)blockIdx.x * blockDim.x + threadIdx.x;
  if (i < n) out[i] = (__bf16)in[i];
}

// ------------------------------------------------- C[M,N] = A[M,K] * Bt[N,K]^T
// A row-major bf16, Bt row-major bf16 (N x K), C f32, optional bias over N.
// Tiles staged into LDS by the Tensor Data Mover (double-buffered, TENSORcnt).
__global__ __launch_bounds__(256) void gemm_bf16_nt(
    const __bf16* __restrict__ A, const __bf16* __restrict__ Bt,
    float* __restrict__ C, const float* __restrict__ bias,
    int M, int N, int K) {
  __shared__ __align__(16) __bf16 sA[2][128 * 32];
  __shared__ __align__(16) __bf16 sB[2][128 * 32];
  const int tid = threadIdx.x;
  const int lane = tid & 31;
  const int wave = tid >> 5;
  const int bm = blockIdx.x * 128, bn = blockIdx.y * 128;
  const int wm = (wave & 1) * 64, wn = (wave >> 1) * 32;

  f32x8 acc[4][2] = {};
  int cur = 0;

#if USE_TDM
  if (wave == 0) {  // one wave drives the DMA pipeline for the workgroup
    tdm_load_tile_bf16(A + (size_t)bm * K, lds_off(&sA[0][0]), 32, 128, (unsigned long long)K);
    tdm_load_tile_bf16(Bt + (size_t)bn * K, lds_off(&sB[0][0]), 32, 128, (unsigned long long)K);
  }
  for (int k0 = 0; k0 < K; k0 += 32) {
    if (wave == 0) {
      if (k0 + 32 < K) {  // prefetch next K-tile into the other buffer
        tdm_load_tile_bf16(A + (size_t)bm * K + (k0 + 32), lds_off(&sA[cur ^ 1][0]),
                           32, 128, (unsigned long long)K);
        tdm_load_tile_bf16(Bt + (size_t)bn * K + (k0 + 32), lds_off(&sB[cur ^ 1][0]),
                           32, 128, (unsigned long long)K);
        // TDM completes in order: <=2 outstanding means current pair is done
        __builtin_amdgcn_s_wait_tensorcnt(2);
      } else {
        __builtin_amdgcn_s_wait_tensorcnt(0);
      }
    }
    __syncthreads();
#else  // fallback: manual VGPR staging
  const int lrow = tid >> 1;         // 0..127
  const int lcol = (tid & 1) * 16;   // 0 or 16 (16 bf16 = 32B per thread)
  for (int k0 = 0; k0 < K; k0 += 32) {
    const uint4* gA = (const uint4*)(A + (size_t)(bm + lrow) * K + k0 + lcol);
    const uint4* gB = (const uint4*)(Bt + (size_t)(bn + lrow) * K + k0 + lcol);
    uint4 a0 = gA[0], a1 = gA[1];
    uint4 b0 = gB[0], b1 = gB[1];
    if (k0 + 32 < K) {
      __builtin_prefetch((const char*)gA + 64, 0, 1);
      __builtin_prefetch((const char*)gB + 64, 0, 1);
    }
    __syncthreads();
    uint4* dA = (uint4*)(&sA[cur][0] + lrow * 32 + lcol);
    dA[0] = a0; dA[1] = a1;
    uint4* dB = (uint4*)(&sB[cur][0] + lrow * 32 + lcol);
    dB[0] = b0; dB[1] = b1;
    __syncthreads();
#endif
    bf16x16 bfr[2];
#pragma unroll
    for (int ni = 0; ni < 2; ++ni)
      bfr[ni] = load_frag_bT(&sB[cur][0] + (wn + ni * 16) * 32, 32);
#pragma unroll
    for (int mi = 0; mi < 4; ++mi) {
      bf16x16 afr = load_frag_a(&sA[cur][0] + (wm + mi * 16) * 32, 32);
#pragma unroll
      for (int ni = 0; ni < 2; ++ni)
        acc[mi][ni] = __builtin_amdgcn_wmma_f32_16x16x32_bf16(
            false, afr, false, bfr[ni], (short)0, acc[mi][ni], false, false);
    }
    __syncthreads();
    cur ^= 1;
  }
  const int colh = lane & 15, rh = (lane >> 4) * 8;
#pragma unroll
  for (int mi = 0; mi < 4; ++mi) {
#pragma unroll
    for (int ni = 0; ni < 2; ++ni) {
      const int row0 = bm + wm + mi * 16 + rh;
      const int col = bn + wn + ni * 16 + colh;
      const float bv = bias ? bias[col] : 0.0f;
#pragma unroll
      for (int r = 0; r < 8; ++r)
        C[(size_t)(row0 + r) * N + col] = acc[mi][ni][r] + bv;
    }
  }
}

// ----------------------------------------------------------------- LSH hashing
// rotated = qk_head @ rot[:,h,:]; buckets = argmax([r, -r]) + h*64
__global__ __launch_bounds__(128) void lsh_hash_kernel(
    const float* __restrict__ qk, const float* __restrict__ rot,
    unsigned* __restrict__ bucket) {
  __shared__ float srot[64 * 128];   // [f][h*32+i], matches rotations layout
  const int bh = blockIdx.x;
  const int b = bh >> 4, hd = bh & 15;
  for (int i = threadIdx.x; i < 64 * 128; i += 128) srot[i] = rot[i];
  __syncthreads();
  const int tok = blockIdx.y * 128 + threadIdx.x;
  const float* qv = qk + ((size_t)b * T_ + tok) * C_ + hd * DH_;
  float v[64];
#pragma unroll
  for (int f = 0; f < 64; ++f) v[f] = qv[f];
  for (int h = 0; h < NHASH_; ++h) {
    float maxv = -1e30f, minv = 1e30f;
    int maxi = 0, mini = 0;
    for (int i = 0; i < 32; ++i) {
      float acc = 0.0f;
#pragma unroll
      for (int f = 0; f < 64; ++f) acc += v[f] * srot[f * 128 + h * 32 + i];
      if (acc > maxv) { maxv = acc; maxi = i; }
      if (acc < minv) { minv = acc; mini = i; }
    }
    // concat([r, -r]) argmax; positives come first so ties go positive
    int bi = (maxv >= -minv) ? maxi : (32 + mini);
    bucket[(size_t)bh * L_ + h * T_ + tok] = (unsigned)(bi + h * (2 * 32));
  }
}

// ------------------------------------------------- stable counting sort per bh
__global__ __launch_bounds__(256) void bucket_sort_kernel(
    const unsigned* __restrict__ bucket, unsigned* __restrict__ sticker,
    unsigned* __restrict__ undo) {
  __shared__ unsigned hist[256];
  __shared__ unsigned offs[256];
  __shared__ unsigned lbuf[32];
  const int bh = blockIdx.x;
  const unsigned* bk = bucket + (size_t)bh * L_;
  unsigned* stk = sticker + (size_t)bh * L_;
  unsigned* und = undo + (size_t)bh * L_;
  const int tid = threadIdx.x;
  hist[tid] = 0;
  __syncthreads();
  for (int i = tid; i < L_; i += 256) atomicAdd(&hist[bk[i]], 1u);
  __syncthreads();
  if (tid == 0) {
    unsigned run = 0;
    for (int i = 0; i < 256; ++i) { offs[i] = run; run += hist[i]; }
  }
  __syncthreads();
  if (tid < 32) {
    // single wave scans in flat-index order -> stable (LDS ops in-order per wave)
    for (int base = 0; base < L_; base += 32) {
      const unsigned b = bk[base + tid];
      lbuf[tid] = b;
      int cnt = 0, tot = 0;
      for (int j = 0; j < 32; ++j) {
        const bool eq = (lbuf[j] == b);
        tot += eq ? 1 : 0;
        cnt += (eq && j < tid) ? 1 : 0;
      }
      const unsigned pos = offs[b] + (unsigned)cnt;
      stk[pos] = (unsigned)(base + tid);
      und[base + tid] = pos;
      if (cnt == tot - 1) offs[b] = pos + 1;
    }
  }
}

// ------------------------------------------------------- chunked LSH attention
__global__ __launch_bounds__(128) void lsh_attn_kernel(
    const float* __restrict__ qk, const float* __restrict__ vv,
    const unsigned* __restrict__ sticker,
    float* __restrict__ so, float* __restrict__ slog) {
  __shared__ __align__(16) __bf16 sq[64 * 64];     // raw queries (chunk c)
  __shared__ __align__(16) __bf16 sk[128 * 64];    // normalized keys, row=kv
  __shared__ __align__(16) __bf16 svT[64 * 128];   // values transposed, row=e
  __shared__ __align__(16) __bf16 sp[64 * 128];    // probs bf16
  __shared__ unsigned qt[64];
  __shared__ unsigned kvt[128];

  const int bh = blockIdx.x, c = blockIdx.y;
  const int b = bh >> 4, hd = bh & 15;
  const int tid = threadIdx.x, lane = tid & 31, wave = tid >> 5;
  const int pc = (c + NCHUNK_ - 1) % NCHUNK_;      // look-one-back with wrap
  const unsigned* stk = sticker + (size_t)bh * L_;

  { // gather: kv row tid (0..127); rows 0..63 double as q rows
    const int j = tid;
    const int slot = (j < 64) ? (c * 64 + j) : (pc * 64 + (j - 64));
    const unsigned tok = stk[slot] % T_;
    kvt[j] = tok;
    const float* kr = qk + ((size_t)b * T_ + tok) * C_ + hd * DH_;
    float tmp[64];
    float ss = 0.0f;
#pragma unroll
    for (int e = 0; e < 64; ++e) { float x = kr[e]; tmp[e] = x; ss += x * x; }
    const float sc = 1.0f / fmaxf(sqrtf(ss), 1e-12f);
#pragma unroll
    for (int e = 0; e < 64; ++e) sk[j * 64 + e] = (__bf16)(tmp[e] * sc);
    const float* vr = vv + ((size_t)b * T_ + tok) * C_ + hd * DH_;
#pragma unroll
    for (int e = 0; e < 64; ++e) svT[e * 128 + j] = (__bf16)vr[e];
    if (j < 64) {
      qt[j] = tok;
#pragma unroll
      for (int e = 0; e < 64; ++e) sq[j * 64 + e] = (__bf16)tmp[e];
    }
  }
  __syncthreads();

  // dots = q(16x64 per wave) . k^T -> 16x128 per wave (8 tiles)
  f32x8 dacc[8] = {};
#pragma unroll
  for (int e0 = 0; e0 < 64; e0 += 32) {
    bf16x16 afr = load_frag_a(sq + (wave * 16) * 64 + e0, 64);
#pragma unroll
    for (int tn = 0; tn < 8; ++tn) {
      bf16x16 bfr = load_frag_bT(sk + (tn * 16) * 64 + e0, 64);
      dacc[tn] = __builtin_amdgcn_wmma_f32_16x16x32_bf16(
          false, afr, false, bfr, (short)0, dacc[tn], false, false);
    }
  }

  const int colh = lane & 15, rh = (lane >> 4) * 8;
  // scale by d^-0.5 and self-mask in registers
#pragma unroll
  for (int tn = 0; tn < 8; ++tn) {
    const unsigned kt = kvt[tn * 16 + colh];
#pragma unroll
    for (int r = 0; r < 8; ++r) {
      const int row = wave * 16 + rh + r;
      const float dv = dacc[tn][r] * 0.125f;
      dacc[tn][r] = (qt[row] == kt) ? -5.0e4f : dv;
    }
  }
  // row logsumexp over 128 cols: 8 tiles local + 16-lane butterfly in the half
  float lse[8];
#pragma unroll
  for (int r = 0; r < 8; ++r) {
    float m = dacc[0][r];
#pragma unroll
    for (int tn = 1; tn < 8; ++tn) m = fmaxf(m, dacc[tn][r]);
#pragma unroll
    for (int d = 1; d < 16; d <<= 1) m = fmaxf(m, __shfl_xor(m, d));
    float s = 0.0f;
#pragma unroll
    for (int tn = 0; tn < 8; ++tn) s += expf(dacc[tn][r] - m);
#pragma unroll
    for (int d = 1; d < 16; d <<= 1) s += __shfl_xor(s, d);
    lse[r] = m + logf(s);
  }
  // probs -> LDS bf16, logits -> global
#pragma unroll
  for (int tn = 0; tn < 8; ++tn) {
#pragma unroll
    for (int r = 0; r < 8; ++r) {
      const int row = wave * 16 + rh + r;
      sp[row * 128 + (tn * 16 + colh)] = (__bf16)expf(dacc[tn][r] - lse[r]);
    }
  }
  if (colh == 0) {
#pragma unroll
    for (int r = 0; r < 8; ++r)
      slog[(size_t)bh * L_ + c * 64 + wave * 16 + rh + r] = lse[r];
  }
  __syncthreads();

  // bo = probs(16x128 per wave) . v(128x64)
  f32x8 oacc[4] = {};
#pragma unroll
  for (int j0 = 0; j0 < 128; j0 += 32) {
    bf16x16 afr = load_frag_a(sp + (wave * 16) * 128 + j0, 128);
#pragma unroll
    for (int tn = 0; tn < 4; ++tn) {
      bf16x16 bfr = load_frag_bT(svT + (tn * 16) * 128 + j0, 128);
      oacc[tn] = __builtin_amdgcn_wmma_f32_16x16x32_bf16(
          false, afr, false, bfr, (short)0, oacc[tn], false, false);
    }
  }
#pragma unroll
  for (int tn = 0; tn < 4; ++tn) {
#pragma unroll
    for (int r = 0; r < 8; ++r) {
      const int row = wave * 16 + rh + r;
      const int col = tn * 16 + colh;
      so[((size_t)bh * L_ + c * 64 + row) * DH_ + col] = oacc[tn][r];
    }
  }
}

// ---------------------------------- unsort + softmax-combine over hash rounds
__global__ __launch_bounds__(256) void combine_kernel(
    const float* __restrict__ so, const float* __restrict__ slog,
    const unsigned* __restrict__ undo, __bf16* __restrict__ ob) {
  const int bh = blockIdx.x;
  const int b = bh >> 4, hd = bh & 15;
  const int tok = blockIdx.y * 256 + threadIdx.x;
  unsigned s[NHASH_];
  float lg[NHASH_];
#pragma unroll
  for (int h = 0; h < NHASH_; ++h) {
    s[h] = undo[(size_t)bh * L_ + h * T_ + tok];
    lg[h] = slog[(size_t)bh * L_ + s[h]];
  }
  float m = lg[0];
#pragma unroll
  for (int h = 1; h < NHASH_; ++h) m = fmaxf(m, lg[h]);
  float w[NHASH_], Z = 0.0f;
#pragma unroll
  for (int h = 0; h < NHASH_; ++h) { w[h] = expf(lg[h] - m); Z += w[h]; }
  const float inv = 1.0f / Z;
  __bf16* op = ob + ((size_t)b * T_ + tok) * C_ + hd * DH_;
  for (int e = 0; e < 64; ++e) {
    float acc = 0.0f;
#pragma unroll
    for (int h = 0; h < NHASH_; ++h)
      acc += w[h] * so[((size_t)bh * L_ + s[h]) * DH_ + e];
    op[e] = (__bf16)(acc * inv);
  }
}

// ----------------------------------------------------------------------------
extern "C" void kernel_launch(void* const* d_in, const int* in_sizes, int n_in,
                              void* d_out, int out_size, void* d_ws, size_t ws_size,
                              hipStream_t stream) {
  (void)in_sizes; (void)n_in; (void)out_size; (void)ws_size;
  const float* x    = (const float*)d_in[0];  // queries
  const float* Wqk  = (const float*)d_in[4];
  const float* Wv   = (const float*)d_in[5];
  const float* Wout = (const float*)d_in[6];
  const float* bout = (const float*)d_in[7];
  const float* rot  = (const float*)d_in[8];

  char* ws = (char*)d_ws;
  size_t off = 0;
  auto take = [&](size_t bytes) -> char* {
    char* p = ws + off;
    off += (bytes + 255) & ~(size_t)255;
    return p;
  };

  __bf16* xb    = (__bf16*)take((size_t)ROWS_ * C_ * 2);
  __bf16* wqkb  = (__bf16*)take((size_t)C_ * C_ * 2);
  __bf16* wvb   = (__bf16*)take((size_t)C_ * C_ * 2);
  __bf16* woutb = (__bf16*)take((size_t)C_ * C_ * 2);
  float* qkf    = (float*)take((size_t)ROWS_ * C_ * 4);
  float* vvf    = (float*)take((size_t)ROWS_ * C_ * 4);
  unsigned* bucket  = (unsigned*)take((size_t)BH_ * L_ * 4);
  unsigned* sticker = (unsigned*)take((size_t)BH_ * L_ * 4);
  unsigned* undo    = (unsigned*)take((size_t)BH_ * L_ * 4);
  float* slog   = (float*)take((size_t)BH_ * L_ * 4);
  float* so     = (float*)take((size_t)BH_ * L_ * DH_ * 4);
  __bf16* ob    = (__bf16*)take((size_t)ROWS_ * C_ * 2);

  const size_t nx = (size_t)ROWS_ * C_;
  const size_t nw = (size_t)C_ * C_;
  f2bf_kernel<<<dim3((unsigned)((nx + 255) / 256)), 256, 0, stream>>>(x, xb, nx);
  f2bf_kernel<<<dim3((unsigned)((nw + 255) / 256)), 256, 0, stream>>>(Wqk, wqkb, nw);
  f2bf_kernel<<<dim3((unsigned)((nw + 255) / 256)), 256, 0, stream>>>(Wv, wvb, nw);
  f2bf_kernel<<<dim3((unsigned)((nw + 255) / 256)), 256, 0, stream>>>(Wout, woutb, nw);

  gemm_bf16_nt<<<dim3(ROWS_ / 128, C_ / 128), 256, 0, stream>>>(
      xb, wqkb, qkf, nullptr, ROWS_, C_, C_);
  gemm_bf16_nt<<<dim3(ROWS_ / 128, C_ / 128), 256, 0, stream>>>(
      xb, wvb, vvf, nullptr, ROWS_, C_, C_);

  lsh_hash_kernel<<<dim3(BH_, T_ / 128), 128, 0, stream>>>(qkf, rot, bucket);
  bucket_sort_kernel<<<dim3(BH_), 256, 0, stream>>>(bucket, sticker, undo);
  lsh_attn_kernel<<<dim3(BH_, NCHUNK_), 128, 0, stream>>>(qkf, vvf, sticker, so, slog);
  combine_kernel<<<dim3(BH_, T_ / 256), 256, 0, stream>>>(so, slog, undo, ob);

  gemm_bf16_nt<<<dim3(ROWS_ / 128, C_ / 128), 256, 0, stream>>>(
      ob, woutb, (float*)d_out, bout, ROWS_, C_, C_);
}
